// TemEnc_78116865179856
// MI455X (gfx1250) — compile-verified, matrix operand order checked by reading, not run on hardware
//
#include <hip/hip_runtime.h>
#include <cmath>

// ---------------- types ----------------
typedef __bf16 bhalf_t;
typedef __attribute__((ext_vector_type(16))) __bf16 v16bf;
typedef __attribute__((ext_vector_type(8)))  __bf16 bf16x8;
typedef __attribute__((ext_vector_type(8)))  float  v8f;

// ---------------- problem constants ----------------
#define B_    8
#define L_    1024
#define CIN_  38
#define D_    512
#define E_    3
#define H_    8
#define DH_   64
#define DFF_  2048
#define TR_   409
#define UN_   615
#define UNP_  624           // UN padded to multiple of 16
#define MENC_ (B_*UNP_)     // 4992
#define MDEC_ (B_*L_)       // 8192

// ---------------- small device helpers ----------------
__device__ __forceinline__ bhalf_t f2bf(float f) {
  unsigned x = __builtin_bit_cast(unsigned, f);
  unsigned r = (x + 0x7FFFu + ((x >> 16) & 1u)) >> 16;       // RNE
  return __builtin_bit_cast(bhalf_t, (unsigned short)r);
}
__device__ __forceinline__ float bf2f(bhalf_t b) {
  unsigned short s = __builtin_bit_cast(unsigned short, b);
  return __builtin_bit_cast(float, ((unsigned)s) << 16);
}
__device__ __forceinline__ float pe_val(int pos, int d) {
  // sinusoidal PE computed analytically (PE table is not a kernel input)
  float i2  = (float)(d & ~1);
  float div = __expf(i2 * (-9.210340371976184f / (float)D_)); // -ln(10000)/D
  float ang = (float)pos * div;
  return (d & 1) ? __cosf(ang) : __sinf(ang);
}
__device__ __forceinline__ float block_sum256(float v) {
  __shared__ float sb[256];
  sb[threadIdx.x] = v; __syncthreads();
  for (int o = 128; o > 0; o >>= 1) {
    if ((int)threadIdx.x < o) sb[threadIdx.x] += sb[threadIdx.x + o];
    __syncthreads();
  }
  float r = sb[0]; __syncthreads();
  return r;
}
__device__ __forceinline__ float block_max256(float v) {
  __shared__ float sb[256];
  sb[threadIdx.x] = v; __syncthreads();
  for (int o = 128; o > 0; o >>= 1) {
    if ((int)threadIdx.x < o) sb[threadIdx.x] = fmaxf(sb[threadIdx.x], sb[threadIdx.x + o]);
    __syncthreads();
  }
  float r = sb[0]; __syncthreads();
  return r;
}

// CDNA5 async memory->LDS (tracked by ASYNCcnt, bypasses VGPRs).
// LDS byte offset = low 32 bits of the generic pointer (LDS aperture layout).
__device__ __forceinline__ unsigned lds_off(const void* p) {
  return (unsigned)(unsigned long long)p;
}
__device__ __forceinline__ void async_b128(void* lds, const void* g) {
  asm volatile("global_load_async_to_lds_b128 %0, %1, off"
               :: "v"(lds_off(lds)), "v"((unsigned long long)g)
               : "memory");
}
__device__ __forceinline__ void wait_async0() {
  asm volatile("s_wait_asynccnt 0x0" ::: "memory");
}

// ---------------- WMMA GEMM ----------------
// C[M,N](f32) = act( A[M,K] @ Bt[N,K]^T + bias[N] ), batched over blockIdx.z.
// Per-z offsets decomposed as z = zo*ZH + zi (for [b,h] style addressing).
// Bt is pre-transposed (row n holds K contiguous) so the B fragment per lane is
// a contiguous-K column slice, matching the ISA per-lane K-striping.
#define BM 128
#define BN 64
#define BK 64     // two 16x16x32 K-slices per stage -> 8 WMMA per barrier pair
#define LDAS 72   // padded LDS row stride (bf16 elems); fragment chunks 16B-aligned
#define LDBS 72

__device__ __forceinline__ v16bf ld_frag(const bhalf_t* base, int rs, int idx, int lh) {
  // lanes 0-15 (lh=0): K = {0..7, 16..23}; lanes 16-31 (lh=1): K = {8..15, 24..31}
  const bhalf_t* p = base + idx * rs + lh * 8;
  bf16x8 lo = *(const bf16x8*)(p);
  bf16x8 hi = *(const bf16x8*)(p + 16);
  v16bf f;
#pragma unroll
  for (int i = 0; i < 8; ++i) { f[i] = lo[i]; f[i + 8] = hi[i]; }
  return f;
}
__device__ __forceinline__ void zero16(bhalf_t* dst) {
  const float4 z = make_float4(0.f, 0.f, 0.f, 0.f);
  *(float4*)(dst)     = z;
  *(float4*)(dst + 8) = z;
}

template <typename TA, int ACT>
__global__ __launch_bounds__(256)
void gemm_wmma(const TA* __restrict__ A,  long ldA, long Ao, long Ai,
               const bhalf_t* __restrict__ Bt, long ldB, long Bo, long Bi,
               const float* __restrict__ bias,
               float* __restrict__ C, long ldC, long Co, long Ci,
               int M, int N, int K, int ZH) {
  __shared__ bhalf_t As[BM * LDAS];
  __shared__ bhalf_t Bs[BN * LDBS];

  const int z  = blockIdx.z;
  const long zo = z / ZH, zi = z % ZH;
  A  += zo * Ao + zi * Ai;
  Bt += zo * Bo + zi * Bi;
  C  += zo * Co + zi * Ci;

  const int m0 = blockIdx.y * BM;
  const int n0 = blockIdx.x * BN;
  const int tid  = threadIdx.x;
  const int lane = tid & 31;
  const int wave = tid >> 5;
  const int wm = (wave & 3) * 32;   // 4 waves over M
  const int wn = (wave >> 2) * 32;  // 2 waves over N
  const int r  = lane & 15;
  const int lh = lane >> 4;

  // staging: A tile 128x64 -> 32 elems/thread (two 16-elem chunks)
  //          B tile  64x64 -> 16 elems/thread (one 16-elem chunk)
  const int arow = tid >> 1;            // 0..127
  const int acol = (tid & 1) * 16;      // chunk 0 at acol, chunk 1 at acol+32
  const int brow = tid >> 2;            // 0..63
  const int bcol = (tid & 3) * 16;

  v8f acc[2][2] = {};

  for (int k0 = 0; k0 < K; k0 += BK) {
    // ---- stage A tile (BM x BK) ----
    {
      const int gm = m0 + arow;
      bhalf_t* dA = &As[arow * LDAS + acol];
      if constexpr (sizeof(TA) == 2) {
        const char* gp = (const char*)(A + (long)gm * ldA + k0 + acol);
        if (gm < M && k0 + acol + 16 <= K) {
          async_b128(dA, gp);  async_b128(dA + 8, gp + 16);
        } else zero16(dA);
        if (gm < M && k0 + acol + 48 <= K) {
          async_b128(dA + 32, gp + 64);  async_b128(dA + 40, gp + 80);
        } else zero16(dA + 32);
      } else {
        if (k0 + BK < K)
          __builtin_prefetch((const void*)(A + (long)gm * ldA + k0 + BK + acol), 0, 3);
#pragma unroll
        for (int cc = 0; cc < 2; ++cc) {
          const int c0 = acol + cc * 32;
          bhalf_t* dst = dA + cc * 32;
          if (gm < M && k0 + c0 + 16 <= K) {
            const float4* s = (const float4*)(A + (long)gm * ldA + k0 + c0);
            float4 v0 = s[0], v1 = s[1], v2 = s[2], v3 = s[3];
            dst[0]  = f2bf(v0.x); dst[1]  = f2bf(v0.y); dst[2]  = f2bf(v0.z); dst[3]  = f2bf(v0.w);
            dst[4]  = f2bf(v1.x); dst[5]  = f2bf(v1.y); dst[6]  = f2bf(v1.z); dst[7]  = f2bf(v1.w);
            dst[8]  = f2bf(v2.x); dst[9]  = f2bf(v2.y); dst[10] = f2bf(v2.z); dst[11] = f2bf(v2.w);
            dst[12] = f2bf(v3.x); dst[13] = f2bf(v3.y); dst[14] = f2bf(v3.z); dst[15] = f2bf(v3.w);
          } else zero16(dst);
        }
      }
    }
    // ---- stage B tile (BN x BK), always bf16 ----
    {
      const int gn = n0 + brow;
      bhalf_t* dB = &Bs[brow * LDBS + bcol];
      const char* gb = (const char*)(Bt + (long)gn * ldB + k0 + bcol);
      if (gn < N && k0 + bcol + 16 <= K) {
        async_b128(dB, gb);  async_b128(dB + 8, gb + 16);
      } else zero16(dB);
    }
    wait_async0();
    __syncthreads();

#pragma unroll
    for (int s = 0; s < 2; ++s) {
      v16bf af[2], bfg[2];
#pragma unroll
      for (int mi = 0; mi < 2; ++mi) af[mi]  = ld_frag(As + s * 32, LDAS, wm + mi * 16 + r, lh);
#pragma unroll
      for (int ni = 0; ni < 2; ++ni) bfg[ni] = ld_frag(Bs + s * 32, LDBS, wn + ni * 16 + r, lh);
#pragma unroll
      for (int mi = 0; mi < 2; ++mi)
#pragma unroll
        for (int ni = 0; ni < 2; ++ni)
          acc[mi][ni] = __builtin_amdgcn_wmma_f32_16x16x32_bf16(
              false, af[mi], false, bfg[ni], (short)0, acc[mi][ni], false, false);
    }
    __syncthreads();
  }

  // ---- epilogue: D-matrix layout (VGPR v: lanes0-15 -> M=v, lanes16-31 -> M=v+8)
#pragma unroll
  for (int mi = 0; mi < 2; ++mi) {
#pragma unroll
    for (int ni = 0; ni < 2; ++ni) {
      const int gn = n0 + wn + ni * 16 + r;
      if (gn >= N) continue;
      const float bb = bias ? bias[gn] : 0.0f;
#pragma unroll
      for (int v = 0; v < 8; ++v) {
        const int gm = m0 + wm + mi * 16 + lh * 8 + v;
        if (gm >= M) continue;
        float val = acc[mi][ni][v] + bb;
        if constexpr (ACT == 1) {
          val = fmaxf(val, 0.0f);
        } else if constexpr (ACT == 2) {
          float t = tanhf(0.7978845608028654f * (val + 0.044715f * val * val * val));
          val = 0.5f * val * (1.0f + t);
        }
        C[(long)gm * ldC + gn] = val;
      }
    }
  }
}

// ---------------- auxiliary kernels ----------------
// transpose + convert weight [K][N] (fp32) -> [N][K] (bf16)
__global__ void wtrans_k(const float* __restrict__ W, bhalf_t* __restrict__ o, int K, int N) {
  long i = (long)blockIdx.x * 256 + threadIdx.x;
  long total = (long)K * N;
  if (i >= total) return;
  long n = i / K, k = i % K;
  o[i] = f2bf(W[k * (long)N + n]);
}

// circular conv(k=3) embedding + positional encoding
__global__ void embed_pe_k(const float* __restrict__ x, const float* __restrict__ w,
                           float* __restrict__ ex) {
  const int bl = blockIdx.x;           // b*L + l
  const int l  = bl & (L_ - 1);
  __shared__ float xs[3 * CIN_];
  if ((int)threadIdx.x < 3 * CIN_) {
    const int t = threadIdx.x / CIN_, c = threadIdx.x % CIN_;
    const int lr = (l + t - 1 + L_) & (L_ - 1);            // wrap pad
    xs[threadIdx.x] = x[(((long)(bl - l) + lr)) * CIN_ + c];
  }
  __syncthreads();
  for (int d = threadIdx.x; d < D_; d += 256) {
    float acc = 0.0f;
#pragma unroll 1
    for (int tc = 0; tc < 3 * CIN_; ++tc) acc += xs[tc] * w[(long)tc * D_ + d];
    ex[(long)bl * D_ + d] = acc + pe_val(l, d);
  }
}

// score[b,l] = 1 / mean_d |ex[l+1]-ex[l]| ; inf at l = L-1
__global__ void diff_score_k(const float* __restrict__ ex, float* __restrict__ score) {
  const int row  = blockIdx.x * 8 + (threadIdx.x >> 5);
  const int lane = threadIdx.x & 31;
  if (row >= B_ * L_) return;
  const int l = row & (L_ - 1);
  float s;
  if (l == L_ - 1) {
    s = __builtin_inff();
  } else {
    const float* a = ex + (long)row * D_;
    const float* c = a + D_;
    float acc = 0.0f;
    for (int d = lane; d < D_; d += 32) acc += fabsf(c[d] - a[d]);
    for (int m = 16; m; m >>= 1) acc += __shfl_xor(acc, m, 32);
    s = (float)D_ / acc;
  }
  if (lane == 0) score[row] = s;
}

// exact top-k split by rank counting + in-block scan (order by position)
__global__ void select_topk_k(const float* __restrict__ score,
                              int* __restrict__ midx, int* __restrict__ uidx) {
  const int b = blockIdx.x;
  __shared__ float sc[L_];
  __shared__ int   scan[L_];
  const int l = threadIdx.x;
  sc[l] = score[(long)b * L_ + l];
  __syncthreads();
  const float mine = sc[l];
  int rank = 0;
  for (int j = 0; j < L_; ++j) {
    const float o = sc[j];
    rank += (o > mine || (o == mine && j < l)) ? 1 : 0;
  }
  const int masked = (rank < TR_) ? 1 : 0;
  const int flagU  = 1 - masked;
  scan[l] = flagU;
  __syncthreads();
  for (int off = 1; off < L_; off <<= 1) {       // Hillis-Steele inclusive scan
    int v = scan[l];
    int add = (l >= off) ? scan[l - off] : 0;
    __syncthreads();
    scan[l] = v + add;
    __syncthreads();
  }
  const int exclU = scan[l] - flagU;
  if (masked) midx[(long)b * TR_ + (l - exclU)] = l;
  else        uidx[(long)b * UN_ + exclU]       = l;
}

// gather unmasked tokens into padded bf16 buffer [B, UNP, D]
__global__ void gather_k(const float* __restrict__ ex, const int* __restrict__ uidx,
                         bhalf_t* __restrict__ xb) {
  const int rr = blockIdx.x;
  const int b = rr / UNP_, j = rr % UNP_;
  bhalf_t* dst = xb + (long)rr * D_;
  if (j < UN_) {
    const int pos = uidx[(long)b * UN_ + j];
    const float* src = ex + ((long)b * L_ + pos) * D_;
    for (int d = threadIdx.x; d < D_; d += 256) dst[d] = f2bf(src[d]);
  } else {
    const bhalf_t z = __builtin_bit_cast(bhalf_t, (unsigned short)0);
    for (int d = threadIdx.x; d < D_; d += 256) dst[d] = z;
  }
}

// scatter encoder output + mask tokens into decoder input [B, L, D] bf16
__global__ void scatter_k(const float* __restrict__ uln, const int* __restrict__ uidx,
                          const int* __restrict__ midx, const float* __restrict__ mtok,
                          bhalf_t* __restrict__ xb) {
  const int rr = blockIdx.x;
  const int b = rr >> 10, j = rr & (L_ - 1);
  if (j < UN_) {
    const int pos = uidx[(long)b * UN_ + j];
    const float* src = uln + ((long)b * UNP_ + j) * D_;
    bhalf_t* dst = xb + ((long)b * L_ + pos) * D_;
    for (int d = threadIdx.x; d < D_; d += 256) dst[d] = f2bf(src[d]);
  } else {
    const int pos = midx[(long)b * TR_ + (j - UN_)];
    bhalf_t* dst = xb + ((long)b * L_ + pos) * D_;
    for (int d = threadIdx.x; d < D_; d += 256) dst[d] = f2bf(mtok[d] + pe_val(pos, d));
  }
}

__global__ void cvt_bf16_k(const float* __restrict__ x, bhalf_t* __restrict__ y, long n) {
  long i = (long)blockIdx.x * 256 + threadIdx.x;
  if (i < n) y[i] = f2bf(x[i]);
}

// V [B, LT, D] fp32 -> Vt [B*H, DH, LT] bf16 (B operand for attn @ V)
__global__ void vtrans_k(const float* __restrict__ V, bhalf_t* __restrict__ Vt, int LT) {
  long i = (long)blockIdx.x * 256 + threadIdx.x;
  const long total = (long)B_ * LT * D_;
  if (i >= total) return;
  const long b = i / ((long)LT * D_);
  const long rem = i % ((long)LT * D_);
  const long t = rem / D_;
  const long c = rem % D_;
  const long h = c >> 6, d = c & 63;
  Vt[(((b * H_ + h) << 6) + d) * (long)LT + t] = f2bf(V[i]);
}

// softmax over rows with valid-length masking; in place; zero padded cols
__global__ void softmax_k(float* __restrict__ S, long ldS, long zS,
                          int valid, int np, float scale) {
  float* p = S + (long)blockIdx.y * zS + (long)blockIdx.x * ldS;
  float mx = -3.0e38f;
  for (int j = threadIdx.x; j < valid; j += 256) mx = fmaxf(mx, p[j]);
  mx = block_max256(mx);
  float sum = 0.0f;
  for (int j = threadIdx.x; j < valid; j += 256) {
    const float e = __expf((p[j] - mx) * scale);
    p[j] = e; sum += e;
  }
  sum = block_sum256(sum);
  const float inv = 1.0f / sum;
  for (int j = threadIdx.x; j < np; j += 256) p[j] = (j < valid) ? p[j] * inv : 0.0f;
}

// layernorm over D=512: in = X*scale (+R), out32 / out16 optional
__global__ void ln_k(const float* __restrict__ X, const float* __restrict__ R, float scale,
                     const float* __restrict__ g, const float* __restrict__ be,
                     float* __restrict__ o32, bhalf_t* __restrict__ o16) {
  const long row = blockIdx.x;
  const float* xr = X + row * D_;
  const int t = threadIdx.x;
  float a = xr[t] * scale;
  float b = xr[t + 256] * scale;
  if (R) { const float* rr = R + row * D_; a += rr[t]; b += rr[t + 256]; }
  const float mean = block_sum256(a + b) * (1.0f / (float)D_);
  const float da = a - mean, db = b - mean;
  const float var = block_sum256(da * da + db * db) * (1.0f / (float)D_);
  const float inv = rsqrtf(var + 1e-5f);
  const float ya = da * inv * g[t] + be[t];
  const float yb = db * inv * g[t + 256] + be[t + 256];
  if (o32) { o32[row * D_ + t] = ya; o32[row * D_ + t + 256] = yb; }
  if (o16) { o16[row * D_ + t] = f2bf(ya); o16[row * D_ + t + 256] = f2bf(yb); }
}

__global__ void sigmoid_k(const float* __restrict__ x, float* __restrict__ y, long n) {
  long i = (long)blockIdx.x * 256 + threadIdx.x;
  if (i < n) y[i] = 1.0f / (1.0f + __expf(-x[i]));
}

// ---------------- host orchestration ----------------
extern "C" void kernel_launch(void* const* d_in, const int* in_sizes, int n_in,
                              void* d_out, int out_size, void* d_ws, size_t ws_size,
                              hipStream_t stream) {
  const float* x     = (const float*)d_in[0];
  const float* embw  = (const float*)d_in[1];
  const float* encWq = (const float*)d_in[2];
  const float* encbq = (const float*)d_in[3];
  const float* encWk = (const float*)d_in[4];
  const float* encbk = (const float*)d_in[5];
  const float* encWv = (const float*)d_in[6];
  const float* encbv = (const float*)d_in[7];
  const float* encWo = (const float*)d_in[8];
  const float* encbo = (const float*)d_in[9];
  const float* eLNg  = (const float*)d_in[10];
  const float* eLNb  = (const float*)d_in[11];
  const float* decWq = (const float*)d_in[12];
  const float* decbq = (const float*)d_in[13];
  const float* decWk = (const float*)d_in[14];
  const float* decbk = (const float*)d_in[15];
  const float* decWv = (const float*)d_in[16];
  const float* decbv = (const float*)d_in[17];
  const float* decWo = (const float*)d_in[18];
  const float* decbo = (const float*)d_in[19];
  const float* c1w   = (const float*)d_in[20];
  const float* c1b   = (const float*)d_in[21];
  const float* c2w   = (const float*)d_in[22];
  const float* c2b   = (const float*)d_in[23];
  const float* ln1g  = (const float*)d_in[24];
  const float* ln1b  = (const float*)d_in[25];
  const float* ln2g  = (const float*)d_in[26];
  const float* ln2b  = (const float*)d_in[27];
  const float* pw1   = (const float*)d_in[28];
  const float* pb1   = (const float*)d_in[29];
  const float* pw2   = (const float*)d_in[30];
  const float* pb2   = (const float*)d_in[31];
  const float* mtok  = (const float*)d_in[32];
  float* out = (float*)d_out;

  const long DD  = (long)D_ * D_;
  const long ATT = (long)B_ * H_ * L_ * L_;

  // ----- workspace carve -----
  char* wsb = (char*)d_ws;
  size_t off = 0;
  auto AL = [&](size_t bytes) -> char* {
    char* p = wsb + off;
    off = (off + bytes + 255) & ~(size_t)255;
    return p;
  };
  bhalf_t* wE  = (bhalf_t*)AL((size_t)E_ * 4 * DD * 2);
  bhalf_t* wD  = (bhalf_t*)AL((size_t)E_ * 4 * DD * 2);
  bhalf_t* c1t = (bhalf_t*)AL((size_t)D_ * DFF_ * 2);
  bhalf_t* c2t = (bhalf_t*)AL((size_t)D_ * DFF_ * 2);
  bhalf_t* p1t = (bhalf_t*)AL((size_t)DD * 2);
  bhalf_t* p2t = (bhalf_t*)AL((size_t)DD * 2);
  float*   ex  = (float*)  AL((size_t)MDEC_ * D_ * 4);
  float*   sc  = (float*)  AL((size_t)B_ * L_ * 4);
  int*     mi_ = (int*)    AL((size_t)B_ * TR_ * 4);
  int*     ui_ = (int*)    AL((size_t)B_ * UN_ * 4);
  bhalf_t* xb  = (bhalf_t*)AL((size_t)MDEC_ * D_ * 2);
  float*   Qf  = (float*)  AL((size_t)MDEC_ * D_ * 4);
  float*   Kf  = (float*)  AL((size_t)MDEC_ * D_ * 4);
  float*   Vf  = (float*)  AL((size_t)MDEC_ * D_ * 4);
  bhalf_t* Qb  = (bhalf_t*)AL((size_t)MDEC_ * D_ * 2);
  bhalf_t* Kb  = (bhalf_t*)AL((size_t)MDEC_ * D_ * 2);
  bhalf_t* Vtb = (bhalf_t*)AL((size_t)B_ * H_ * DH_ * L_ * 2);
  float*   ctxf= (float*)  AL((size_t)MDEC_ * D_ * 4);
  bhalf_t* ctxb= (bhalf_t*)AL((size_t)MDEC_ * D_ * 2);
  float*   yf  = (float*)  AL((size_t)MDEC_ * D_ * 4);
  float*   lnf = (float*)  AL((size_t)MDEC_ * D_ * 4);
  // phase-exclusive region: encoder scores  vs  FFN intermediate
  size_t bigBytes = (size_t)MDEC_ * DFF_ * 4 + (size_t)MDEC_ * DFF_ * 2; // 100.7 MB
  size_t encBytes = (size_t)B_ * H_ * UNP_ * UNP_ * 4;                   //  99.7 MB
  if (encBytes > bigBytes) bigBytes = encBytes;
  char* big = AL(bigBytes);
  float*   encS = (float*)big;
  float*   ff1  = (float*)big;
  bhalf_t* g1b  = (bhalf_t*)(big + (size_t)MDEC_ * DFF_ * 4);

  auto gridOf = [](long n) { return dim3((unsigned)((n + 255) / 256)); };

  // ----- weight prep: fp32 [K][N] -> bf16 [N][K] -----
  for (int i = 0; i < E_; ++i) {
    wtrans_k<<<gridOf(DD), 256, 0, stream>>>(encWq + (long)i * DD, wE + ((long)i*4+0)*DD, D_, D_);
    wtrans_k<<<gridOf(DD), 256, 0, stream>>>(encWk + (long)i * DD, wE + ((long)i*4+1)*DD, D_, D_);
    wtrans_k<<<gridOf(DD), 256, 0, stream>>>(encWv + (long)i * DD, wE + ((long)i*4+2)*DD, D_, D_);
    wtrans_k<<<gridOf(DD), 256, 0, stream>>>(encWo + (long)i * DD, wE + ((long)i*4+3)*DD, D_, D_);
    wtrans_k<<<gridOf(DD), 256, 0, stream>>>(decWq + (long)i * DD, wD + ((long)i*4+0)*DD, D_, D_);
    wtrans_k<<<gridOf(DD), 256, 0, stream>>>(decWk + (long)i * DD, wD + ((long)i*4+1)*DD, D_, D_);
    wtrans_k<<<gridOf(DD), 256, 0, stream>>>(decWv + (long)i * DD, wD + ((long)i*4+2)*DD, D_, D_);
    wtrans_k<<<gridOf(DD), 256, 0, stream>>>(decWo + (long)i * DD, wD + ((long)i*4+3)*DD, D_, D_);
  }
  wtrans_k<<<gridOf((long)D_*DFF_), 256, 0, stream>>>(c1w, c1t, D_, DFF_);
  wtrans_k<<<gridOf((long)D_*DFF_), 256, 0, stream>>>(c2w, c2t, DFF_, D_);
  wtrans_k<<<gridOf(DD), 256, 0, stream>>>(pw1, p1t, D_, D_);
  wtrans_k<<<gridOf(DD), 256, 0, stream>>>(pw2, p2t, D_, D_);

  // ----- embedding + PE, score, top-k split, gather -----
  embed_pe_k<<<B_ * L_, 256, 0, stream>>>(x, embw, ex);
  diff_score_k<<<(B_ * L_) / 8, 256, 0, stream>>>(ex, sc);
  select_topk_k<<<B_, 1024, 0, stream>>>(sc, mi_, ui_);
  gather_k<<<MENC_, 256, 0, stream>>>(ex, ui_, xb);

  const dim3 gP((D_ + 63) / 64, (MENC_ + 127) / 128, 1);     // enc projections
  // ----- encoder (3 x MHA, no residual) -----
  for (int i = 0; i < E_; ++i) {
    const bhalf_t* wq = wE + ((long)i*4+0)*DD;
    const bhalf_t* wk = wE + ((long)i*4+1)*DD;
    const bhalf_t* wv = wE + ((long)i*4+2)*DD;
    const bhalf_t* wo = wE + ((long)i*4+3)*DD;
    gemm_wmma<bhalf_t,0><<<gP,256,0,stream>>>(xb,(long)D_,0,0, wq,(long)D_,0,0, encbq+(long)i*D_, Qf,(long)D_,0,0, MENC_,D_,D_,1);
    gemm_wmma<bhalf_t,0><<<gP,256,0,stream>>>(xb,(long)D_,0,0, wk,(long)D_,0,0, encbk+(long)i*D_, Kf,(long)D_,0,0, MENC_,D_,D_,1);
    gemm_wmma<bhalf_t,0><<<gP,256,0,stream>>>(xb,(long)D_,0,0, wv,(long)D_,0,0, encbv+(long)i*D_, Vf,(long)D_,0,0, MENC_,D_,D_,1);
    cvt_bf16_k<<<gridOf((long)MENC_*D_),256,0,stream>>>(Qf, Qb, (long)MENC_*D_);
    cvt_bf16_k<<<gridOf((long)MENC_*D_),256,0,stream>>>(Kf, Kb, (long)MENC_*D_);
    vtrans_k<<<gridOf((long)B_*UNP_*D_),256,0,stream>>>(Vf, Vtb, UNP_);
    // scores = Q @ K^T per (b,h)
    gemm_wmma<bhalf_t,0><<<dim3((UNP_+63)/64,(UNP_+127)/128,B_*H_),256,0,stream>>>(
        Qb,(long)D_,(long)UNP_*D_,(long)DH_,
        Kb,(long)D_,(long)UNP_*D_,(long)DH_, nullptr,
        encS,(long)UNP_,(long)H_*UNP_*UNP_,(long)UNP_*UNP_, UNP_,UNP_,DH_,H_);
    softmax_k<<<dim3(UNP_, B_*H_),256,0,stream>>>(encS,(long)UNP_,(long)UNP_*UNP_, UN_, UNP_, 0.125f);
    // ctx = attn @ V
    gemm_wmma<float,0><<<dim3(1,(UNP_+127)/128,B_*H_),256,0,stream>>>(
        encS,(long)UNP_,(long)H_*UNP_*UNP_,(long)UNP_*UNP_,
        Vtb,(long)UNP_,(long)H_*DH_*UNP_,(long)DH_*UNP_, nullptr,
        ctxf,(long)D_,(long)UNP_*D_,(long)DH_, UNP_,DH_,UNP_,H_);
    cvt_bf16_k<<<gridOf((long)MENC_*D_),256,0,stream>>>(ctxf, ctxb, (long)MENC_*D_);
    gemm_wmma<bhalf_t,0><<<gP,256,0,stream>>>(ctxb,(long)D_,0,0, wo,(long)D_,0,0, encbo+(long)i*D_, yf,(long)D_,0,0, MENC_,D_,D_,1);
    cvt_bf16_k<<<gridOf((long)MENC_*D_),256,0,stream>>>(yf, xb, (long)MENC_*D_);
  }
  ln_k<<<MENC_,256,0,stream>>>(yf, nullptr, 1.0f, eLNg, eLNb, lnf, (bhalf_t*)nullptr);
  scatter_k<<<B_*L_,256,0,stream>>>(lnf, ui_, mi_, mtok, xb);

  // ----- decoder (3 x MHA, attention probs -> d_out) -----
  const dim3 gPD((D_ + 63) / 64, (MDEC_ + 127) / 128, 1);
  for (int i = 0; i < E_; ++i) {
    const bhalf_t* wq = wD + ((long)i*4+0)*DD;
    const bhalf_t* wk = wD + ((long)i*4+1)*DD;
    const bhalf_t* wv = wD + ((long)i*4+2)*DD;
    const bhalf_t* wo = wD + ((long)i*4+3)*DD;
    gemm_wmma<bhalf_t,0><<<gPD,256,0,stream>>>(xb,(long)D_,0,0, wq,(long)D_,0,0, decbq+(long)i*D_, Qf,(long)D_,0,0, MDEC_,D_,D_,1);
    gemm_wmma<bhalf_t,0><<<gPD,256,0,stream>>>(xb,(long)D_,0,0, wk,(long)D_,0,0, decbk+(long)i*D_, Kf,(long)D_,0,0, MDEC_,D_,D_,1);
    gemm_wmma<bhalf_t,0><<<gPD,256,0,stream>>>(xb,(long)D_,0,0, wv,(long)D_,0,0, decbv+(long)i*D_, Vf,(long)D_,0,0, MDEC_,D_,D_,1);
    cvt_bf16_k<<<gridOf((long)MDEC_*D_),256,0,stream>>>(Qf, Qb, (long)MDEC_*D_);
    cvt_bf16_k<<<gridOf((long)MDEC_*D_),256,0,stream>>>(Kf, Kb, (long)MDEC_*D_);
    vtrans_k<<<gridOf((long)B_*L_*D_),256,0,stream>>>(Vf, Vtb, L_);
    float* att = out + (long)i * ATT;                        // [B,H,L,L]
    gemm_wmma<bhalf_t,0><<<dim3(L_/64, L_/128, B_*H_),256,0,stream>>>(
        Qb,(long)D_,(long)L_*D_,(long)DH_,
        Kb,(long)D_,(long)L_*D_,(long)DH_, nullptr,
        att,(long)L_,(long)H_*L_*L_,(long)L_*L_, L_,L_,DH_,H_);
    softmax_k<<<dim3(L_, B_*H_),256,0,stream>>>(att,(long)L_,(long)L_*L_, L_, L_, 0.125f);
    gemm_wmma<float,0><<<dim3(1, L_/128, B_*H_),256,0,stream>>>(
        att,(long)L_,(long)H_*L_*L_,(long)L_*L_,
        Vtb,(long)L_,(long)H_*DH_*L_,(long)DH_*L_, nullptr,
        ctxf,(long)D_,(long)L_*D_,(long)DH_, L_,DH_,L_,H_);
    cvt_bf16_k<<<gridOf((long)MDEC_*D_),256,0,stream>>>(ctxf, ctxb, (long)MDEC_*D_);
    gemm_wmma<bhalf_t,0><<<gPD,256,0,stream>>>(ctxb,(long)D_,0,0, wo,(long)D_,0,0, decbo+(long)i*D_, yf,(long)D_,0,0, MDEC_,D_,D_,1);
    cvt_bf16_k<<<gridOf((long)MDEC_*D_),256,0,stream>>>(yf, xb, (long)MDEC_*D_);
  }

  // ----- dx = LN1(2*dx); FFN; LN2(dx + y); projection head -----
  ln_k<<<MDEC_,256,0,stream>>>(yf, nullptr, 2.0f, ln1g, ln1b, lnf, xb);
  gemm_wmma<bhalf_t,1><<<dim3(DFF_/64, MDEC_/128, 1),256,0,stream>>>(
      xb,(long)D_,0,0, c1t,(long)D_,0,0, c1b, ff1,(long)DFF_,0,0, MDEC_,DFF_,D_,1);
  cvt_bf16_k<<<gridOf((long)MDEC_*DFF_),256,0,stream>>>(ff1, g1b, (long)MDEC_*DFF_);
  gemm_wmma<bhalf_t,0><<<gPD,256,0,stream>>>(
      g1b,(long)DFF_,0,0, c2t,(long)DFF_,0,0, c2b, yf,(long)D_,0,0, MDEC_,D_,DFF_,1);
  ln_k<<<MDEC_,256,0,stream>>>(lnf, yf, 1.0f, ln2g, ln2b, (float*)nullptr, xb);
  gemm_wmma<bhalf_t,2><<<gPD,256,0,stream>>>(
      xb,(long)D_,0,0, p1t,(long)D_,0,0, pb1, Qf,(long)D_,0,0, MDEC_,D_,D_,1);
  cvt_bf16_k<<<gridOf((long)MDEC_*D_),256,0,stream>>>(Qf, Qb, (long)MDEC_*D_);
  gemm_wmma<bhalf_t,0><<<gPD,256,0,stream>>>(
      Qb,(long)D_,0,0, p2t,(long)D_,0,0, pb2, yf,(long)D_,0,0, MDEC_,D_,D_,1);
  sigmoid_k<<<gridOf((long)MDEC_*D_),256,0,stream>>>(yf, out + 3 * ATT, (long)MDEC_*D_);
}
